// ARDSAS_75359496175652
// MI455X (gfx1250) — matrix operand order repaired and checked by reading, not compiled, not run on hardware
//
#include <hip/hip_runtime.h>
#include <cstdint>
#include <cstddef>

// ---------------------------------------------------------------------------
// ARDSAS forward for MI455X (gfx1250, wave32, WMMA).
// All GEMM-shaped math runs on v_wmma_f32_16x16x32_bf16.
// B operands (weights, attention V) are pre-packed fragment-major so every
// WMMA operand fetch is two global_load_b128 (no strided u16 loads).
// GEMM waves own 16x64 output (4 independent WMMAs per K-step -> hazard-free
// back-to-back issue, 4x A-fragment reuse).
// Workspace requirement: ~660 MB (kv4 211MB + neg4 210MB + activations).
// ---------------------------------------------------------------------------

#define N_   128
#define L_   200
#define E_   128
#define NB_  2
#define H_   2
#define HD_  64
#define T1_  (N_ * (L_ + 1))   // 25728 tokens on the kv path (multiple of 16)
#define T0_  (N_ * L_)         // 25600 tokens on the seq path (multiple of 16)
#define LP_  224               // key axis padded to a multiple of 32 for WMMA

typedef __attribute__((ext_vector_type(16))) __bf16 v16bf;
typedef __attribute__((ext_vector_type(8)))  float  v8f;

#define GF_BIAS 1
#define GF_RELU 2
#define GF_RES  4
#define GF_BF16 8

__device__ inline __bf16 tobf(float f) {
  unsigned u = __builtin_bit_cast(unsigned, f);
  u += 0x7fffu + ((u >> 16) & 1u);           // round-to-nearest-even
  unsigned short h = (unsigned short)(u >> 16);
  return __builtin_bit_cast(__bf16, h);
}

__device__ inline v8f wmma_bf16(v16bf a, v16bf b, v8f c) {
  return __builtin_amdgcn_wmma_f32_16x16x32_bf16(false, a, false, b,
                                                 (short)0, c, false, false);
}

// 16-bit A-fragment K offset for element pair p in lane half hi
// (lanes 0-15: K=0..15; lanes 16-31: K=16..31)
__device__ inline int a_koff(int p, int hi) {
  return hi * 16 + 2 * p;
}

__device__ inline v16bf load_a_frag(const __bf16* __restrict__ Ar, int hi) {
  v16bf af;
#pragma unroll
  for (int p = 0; p < 8; ++p) {
    int kb = a_koff(p, hi);
    af[2 * p]     = Ar[kb];
    af[2 * p + 1] = Ar[kb + 1];
  }
  return af;
}

__device__ inline v16bf load_packed_frag(const __bf16* __restrict__ Bp) {
  v16bf bf;
#pragma unroll
  for (int j = 0; j < 16; ++j) bf[j] = Bp[j];  // contiguous: 2x b128
  return bf;
}

__device__ inline float blockReduce128(float v, float* red) {
  int tid = threadIdx.x;
  red[tid] = v; __syncthreads();
#pragma unroll
  for (int s = 64; s > 0; s >>= 1) {
    if (tid < s) red[tid] += red[tid + s];
    __syncthreads();
  }
  float r = red[0]; __syncthreads();
  return r;
}

// ---------------------------------------------------------------------------
// Generic bf16 WMMA GEMM on packed B:
//   C(M,N) = A(M,K) @ B(K,N)  [+bias][+resid][relu][bf16 dup]
// One wave owns a 16x64 C stripe (4 column tiles, 4 independent WMMAs/K-step).
// Requires M%16==0, N%64==0, K%32==0 (always true here).
// Packed B layout: idx = ((nt*(K/32) + kt)*32 + lane)*16 + j
//   where element = B[kt*32 + (lane>>4)*16 + j][nt*16 + (lane&15)].
// ---------------------------------------------------------------------------
__global__ __launch_bounds__(128) void wmma_gemm(
    const __bf16* __restrict__ A, int lda,
    const __bf16* __restrict__ Bp,
    float* __restrict__ C, int ldc,
    const float* __restrict__ bias,
    const float* __restrict__ resid, int ldres,
    __bf16* __restrict__ Cbf,
    int M, int Nn, int K, int flags)
{
  int wid = blockIdx.x * 4 + (threadIdx.x >> 5);
  int ntG = Nn >> 6;                       // groups of 4 column tiles
  int tM = wid / ntG, tG = wid % ntG;
  if (tM * 16 >= M) return;
  int lane = threadIdx.x & 31;
  int ml = lane & 15, hi = lane >> 4;
  int kt32 = K >> 5;
  size_t tileStride = (size_t)kt32 * 512;  // packed advance per column tile
  const __bf16* Arow = A + (size_t)(tM * 16 + ml) * lda;
  const __bf16* Bg = Bp + (((size_t)(tG * 4) * kt32) * 32 + lane) * 16;
  v8f acc0 = {}, acc1 = {}, acc2 = {}, acc3 = {};
  for (int kk = 0; kk < K; kk += 32) {
    v16bf af = load_a_frag(Arow + kk, hi);
    const __bf16* Bk = Bg + (size_t)(kk >> 5) * 512;
    v16bf b0 = load_packed_frag(Bk);
    v16bf b1 = load_packed_frag(Bk + tileStride);
    v16bf b2 = load_packed_frag(Bk + 2 * tileStride);
    v16bf b3 = load_packed_frag(Bk + 3 * tileStride);
    acc0 = wmma_bf16(af, b0, acc0);
    acc1 = wmma_bf16(af, b1, acc1);
    acc2 = wmma_bf16(af, b2, acc2);
    acc3 = wmma_bf16(af, b3, acc3);
  }
  v8f accs[4] = {acc0, acc1, acc2, acc3};
#pragma unroll
  for (int c = 0; c < 4; ++c) {
    int col = (tG * 4 + c) * 16 + ml;
    float bv = (flags & GF_BIAS) ? bias[col] : 0.f;
#pragma unroll
    for (int r = 0; r < 8; ++r) {
      int m = tM * 16 + hi * 8 + r;
      float v = accs[c][r] + bv;
      if (flags & GF_RES)  v += resid[(size_t)m * ldres + col];
      if (flags & GF_RELU) v = v > 0.f ? v : 0.f;
      C[(size_t)m * ldc + col] = v;
      if (flags & GF_BF16) Cbf[(size_t)m * ldc + col] = tobf(v);
    }
  }
}

// ---------------------------------------------------------------------------
// Embedding gathers + masks + bf16 copies
// ---------------------------------------------------------------------------
__global__ void gather_kernel(
    const float* __restrict__ item_emb,
    const int* __restrict__ log_seqs, const int* __restrict__ pos_seqs,
    const int* __restrict__ neg_seqs,
    float* __restrict__ kv_save, __bf16* __restrict__ kv_bf,
    float* __restrict__ validE, float* __restrict__ validL,
    float* __restrict__ neg_emb, __bf16* __restrict__ neg_bf)
{
  int idx = blockIdx.x * blockDim.x + threadIdx.x;
  if (idx >= T1_ * E_) return;
  int e = idx & (E_ - 1);
  int t = idx >> 7;
  int n = t / (L_ + 1), l = t % (L_ + 1);
  int id; float vm;
  if (l < L_) { id = log_seqs[n * L_ + l]; vm = id ? 1.f : 0.f; }
  else        { id = pos_seqs[n * L_ + L_ - 1];
                vm = log_seqs[n * L_ + L_ - 1] ? 1.f : 0.f; }
  float v = item_emb[(size_t)id * E_ + e] * vm;
  kv_save[idx] = v; kv_bf[idx] = tobf(v);
  if (e == 0) { validE[t] = vm; if (l < L_) validL[n * L_ + l] = vm; }
  if (l < L_) {
    int nid = neg_seqs[n * L_ + l];
    float nv = item_emb[(size_t)nid * E_ + e] * vm;
    size_t o = ((size_t)(n * L_ + l)) * E_ + e;
    neg_emb[o] = nv; neg_bf[o] = tobf(nv);
  }
}

// W_asp (16,E,E) -> packed bf16 B (K=E, N=16*E) for the fused aspect GEMM
__global__ void build_wasp_packed(const float* __restrict__ W_asp,
                                  __bf16* __restrict__ Wp)
{
  int idx = blockIdx.x * blockDim.x + threadIdx.x;
  if (idx >= 2048 * E_) return;
  int j    = idx & 15;
  int lane = (idx >> 4) & 31;
  int kt   = (idx >> 9) & 3;               // K=128 -> 4 k-tiles
  int nt   = idx >> 11;                    // 128 n-tiles
  int k = kt * 32 + (lane >> 4) * 16 + j;
  int n = nt * 16 + (lane & 15);
  int a = n >> 7, o = n & 127;
  Wp[idx] = tobf(W_asp[((size_t)a * E_ + o) * E_ + k]);
}

// mean over aspects of W_asp -> packed (K=E, N=E); cent: temp.mean == x@Wmean^T
__global__ void build_wmean_packed(const float* __restrict__ W_asp,
                                   __bf16* __restrict__ Wmp)
{
  int idx = blockIdx.x * blockDim.x + threadIdx.x;
  if (idx >= E_ * E_) return;
  int j    = idx & 15;
  int lane = (idx >> 4) & 31;
  int kt   = (idx >> 9) & 3;
  int nt   = idx >> 11;
  int k = kt * 32 + (lane >> 4) * 16 + j;
  int o = nt * 16 + (lane & 15);
  float m = 0.f;
  for (int a = 0; a < 16; ++a) m += W_asp[((size_t)a * E_ + o) * E_ + k];
  Wmp[idx] = tobf(m * (1.f / 16.f));
}

// W (O,I) fp32 -> packed bf16 B with B[k][n] = W[n][k]  (N=O, K=I)
__global__ void pack_weight(const float* __restrict__ W,
                            __bf16* __restrict__ Wp, int O, int I)
{
  int idx = blockIdx.x * blockDim.x + threadIdx.x;
  if (idx >= O * I) return;
  int j    = idx & 15;
  int lane = (idx >> 4) & 31;
  int kt   = (idx >> 9) % (I >> 5);
  int nt   = idx / ((I >> 5) << 9);
  int k = kt * 32 + (lane >> 4) * 16 + j;
  int n = nt * 16 + (lane & 15);
  Wp[idx] = tobf(W[(size_t)n * I + k]);
}

// repack attention V (from kvp_bf, row stride 256) into fragment-major layout:
// Vp[(((n*H+h)*(LP/32)+k32)*4 + ot)*512 + lane*16 + j]
//   = V[key = k32*32+(lane>>4)*16+j][d = ot*16+(lane&15)]   (0 beyond L)
__global__ void pack_v_kernel(const __bf16* __restrict__ kvp,
                              __bf16* __restrict__ Vp)
{
  int idx = blockIdx.x * blockDim.x + threadIdx.x;
  if (idx >= N_ * H_ * (LP_ / 32) * 2048) return;
  int j    = idx & 15;
  int lane = (idx >> 4) & 31;
  int ot   = (idx >> 9) & 3;
  int k32  = (idx >> 11) % (LP_ / 32);
  int nh   = idx / ((LP_ / 32) << 11);
  int h = nh & (H_ - 1), n = nh >> 1;
  int key = k32 * 32 + (lane >> 4) * 16 + j;
  __bf16 v = tobf(0.f);
  if (key < L_)
    v = kvp[((size_t)(n * L_ + key)) * 256 + 128 + h * HD_ + ot * 16 + (lane & 15)];
  Vp[idx] = v;
}

// residual add + per-aspect L2 normalize + mask; s = sum over aspects;
// optional cent-loss contribution ||x - tmean|| * valid
__global__ __launch_bounds__(128) void aspect_post(
    const float* __restrict__ x, float* __restrict__ a4,
    const float* __restrict__ valid, const float* __restrict__ tmean,
    float* __restrict__ s_out, float* __restrict__ cent_acc, int has_cent)
{
  __shared__ float red[128];
  int t = blockIdx.x, e = threadIdx.x;
  float xv = x[(size_t)t * E_ + e];
  float vm = valid[t];
  float* A = a4 + (size_t)t * 2048;
  float s_e = 0.f;
  for (int a = 0; a < 16; ++a) {
    float v = A[a * E_ + e] + xv;
    float nrm = blockReduce128(v * v, red);
    v = v / (sqrtf(nrm) + 1e-8f) * vm;
    A[a * E_ + e] = v;
    s_e += v;
  }
  if (s_out) s_out[(size_t)t * E_ + e] = s_e;
  if (has_cent) {
    float d = (xv - tmean[(size_t)t * E_ + e]) * vm;
    float ss = blockReduce128(d * d, red);
    if (e == 0) atomicAdd(cent_acc, sqrtf(ss));
  }
}

// per-token 16x16 aspect gram via WMMA + InfoNCE-style loss accumulation
__global__ __launch_bounds__(256) void ind_kernel(
    const float* __restrict__ kv4, const float* __restrict__ validE,
    float* __restrict__ ind_acc)
{
  __shared__ __bf16 nx[8][16 * E_];
  __shared__ float  nr[8][16];
  int w = threadIdx.x >> 5, lane = threadIdx.x & 31;
  int t = blockIdx.x * 8 + w;                 // T1_ is a multiple of 8 waves
  bool act = validE[t] > 0.f;                 // uniform per wave
  if (act) {
    for (int r = 0; r < 16; ++r) {
      const float* row = kv4 + (size_t)t * 2048 + r * E_;
      float ss = 0.f;
      for (int e = lane; e < E_; e += 32) { float v = row[e]; ss += v * v; }
#pragma unroll
      for (int m = 16; m >= 1; m >>= 1) ss += __shfl_xor(ss, m, 32);
      if (lane == 0) nr[w][r] = sqrtf(ss) + 1e-8f;
    }
  }
  __syncthreads();
  if (act) {
    for (int idx = lane; idx < 16 * E_; idx += 32)
      nx[w][idx] = tobf(kv4[(size_t)t * 2048 + idx] / nr[w][idx >> 7]);
  }
  __syncthreads();
  if (act) {
    int ml = lane & 15, hi = lane >> 4;
    v8f acc = {};
    for (int kk = 0; kk < E_; kk += 32) {
      v16bf af = load_a_frag(&nx[w][ml * E_ + kk], hi);
      v16bf bf;                               // B = A^T of same matrix
#pragma unroll
      for (int j = 0; j < 16; ++j)
        bf[j] = nx[w][ml * E_ + kk + hi * 16 + j];
      acc = wmma_bf16(af, bf, acc);
    }
    const float invT = 5.f;                   // 1 / TEMP
    float part = 0.f;
#pragma unroll
    for (int r = 0; r < 8; ++r) {
      int rowi = hi * 8 + r;
      float sum = __expf(acc[r] * invT);
#pragma unroll
      for (int m = 1; m < 16; m <<= 1) sum += __shfl_xor(sum, m, 32);
      float diag = __shfl(acc[r], hi * 16 + rowi, 32);   // pos_s == ttl diag
      float mi = -(diag * invT - __logf(sum));
      if (ml == 0) part += mi;
    }
    float other = __shfl(part, 16, 32);
    if (lane == 0) atomicAdd(ind_acc, part + other);
  }
}

// exclusive cumsum of s over L+1 per sequence -> q; also nvalid
__global__ __launch_bounds__(128) void cumsum_kernel(
    const float* __restrict__ s, const float* __restrict__ validE,
    float* __restrict__ q, float* __restrict__ nvalid)
{
  int n = blockIdx.x, e = threadIdx.x;
  float run = 0.f, cnt = 0.f;
  for (int l = 0; l <= L_; ++l) {
    size_t t = (size_t)n * (L_ + 1) + l;
    float vm = validE[t];
    q[t * E_ + e] = run / (16.f * cnt + 1e-8f) * vm;
    run += s[t * E_ + e];
    cnt += vm;
  }
  if (e == 0) atomicAdd(nvalid, cnt);
}

// aspect-gate softmax; mode 0: re_all + aspect-mean over kv4 (T1 tokens)
// mode 1: neg_t = gate(q2, neg4) + neg4.mean(2) (T0 tokens, q shifted+masked)
__global__ __launch_bounds__(128) void gate_kernel(
    const float* __restrict__ q, const float* __restrict__ a4,
    const float* __restrict__ w_weight, const float* __restrict__ w_bias,
    const float* __restrict__ validE,
    float* __restrict__ out, float* __restrict__ meanout, int mode)
{
  __shared__ float red[128];
  __shared__ float lg[16];
  int t = blockIdx.x, e = threadIdx.x;
  size_t qi; float vm = 1.f;
  if (mode == 0) qi = (size_t)t * E_;
  else {
    int n = t / L_, l = t % L_;
    qi = ((size_t)(n * (L_ + 1) + l + 1)) * E_;
    vm = validE[n * (L_ + 1) + l];
  }
  float qv = q[qi + e] * (mode ? vm : 1.f);
  float dq = blockReduce128(qv * w_weight[e], red);
  const float* A = a4 + (size_t)t * 2048;
  float av[16]; float mean_e = 0.f;
  float wa = w_weight[E_ + e];
#pragma unroll
  for (int a = 0; a < 16; ++a) {
    av[a] = A[a * E_ + e];
    float da = blockReduce128(av[a] * wa, red);
    if (e == 0) lg[a] = dq + da + w_bias[0];
    mean_e += av[a];
  }
  __syncthreads();
  float mx = -1e30f;
#pragma unroll
  for (int a = 0; a < 16; ++a) mx = fmaxf(mx, lg[a]);
  float sum = 0.f; float wsm[16];
#pragma unroll
  for (int a = 0; a < 16; ++a) { wsm[a] = __expf(lg[a] - mx); sum += wsm[a]; }
  float inv = 1.f / sum;
  float o = 0.f;
#pragma unroll
  for (int a = 0; a < 16; ++a) o += av[a] * wsm[a] * inv;
  mean_e *= (1.f / 16.f);
  if (mode == 0) {
    out[(size_t)t * E_ + e] = o;
    meanout[(size_t)t * E_ + e] = mean_e;
  } else {
    out[(size_t)t * E_ + e] = o + mean_e;
  }
}

// seqs = (re_all+mean)*sqrt(E) + pos_emb, masked; pos_t from shifted tokens
__global__ __launch_bounds__(256) void build_seq(
    const float* __restrict__ re_all, const float* __restrict__ kvmean,
    const float* __restrict__ validE, const float* __restrict__ pos_emb,
    float* __restrict__ seqs, __bf16* __restrict__ seqs_bf,
    float* __restrict__ pos_t)
{
  int idx = blockIdx.x * blockDim.x + threadIdx.x;
  if (idx >= T0_ * E_) return;
  int e = idx & (E_ - 1);
  int t0 = idx >> 7;
  int n = t0 / L_, l = t0 % L_;
  size_t t1 = (size_t)n * (L_ + 1) + l;
  float vm = validE[t1];
  float re = re_all[t1 * E_ + e] + kvmean[t1 * E_ + e];
  float sv = (re * 11.3137084989847604f + pos_emb[(size_t)l * E_ + e]) * vm;
  seqs[idx] = sv; seqs_bf[idx] = tobf(sv);
  pos_t[idx] = (re_all[(t1 + 1) * E_ + e] + kvmean[(t1 + 1) * E_ + e]) * vm;
}

__global__ __launch_bounds__(256) void mask_bf_kernel(
    float* __restrict__ seqs, __bf16* __restrict__ sbf,
    const float* __restrict__ validE)
{
  int idx = blockIdx.x * blockDim.x + threadIdx.x;
  if (idx >= T0_ * E_) return;
  int t0 = idx >> 7; int n = t0 / L_, l = t0 % L_;
  float v = seqs[idx] * validE[(size_t)n * (L_ + 1) + l];
  seqs[idx] = v; sbf[idx] = tobf(v);
}

__global__ __launch_bounds__(128) void ln_kernel(
    const float* __restrict__ x, const float* __restrict__ g,
    const float* __restrict__ b, float* __restrict__ y,
    __bf16* __restrict__ ybf)
{
  __shared__ float red[128];
  int t = blockIdx.x, e = threadIdx.x;
  float v = x[(size_t)t * E_ + e];
  float m = blockReduce128(v, red) * (1.f / E_);
  float d = v - m;
  float var = blockReduce128(d * d, red) * (1.f / E_);
  float o = d * rsqrtf(var + 1e-8f) * g[e] + b[e];
  y[(size_t)t * E_ + e] = o;
  if (ybf) ybf[(size_t)t * E_ + e] = tobf(o);
}

// ---------------------------------------------------------------------------
// Causal attention: scores via WMMA (K=64), softmax in LDS, P@V via WMMA
// over LP_=224 padded keys using pre-packed V fragments.
// One wave per (n, head, 16-row q tile); 2 waves per block.
// ---------------------------------------------------------------------------
__global__ __launch_bounds__(64) void attn_kernel(
    const __bf16* __restrict__ qp, const __bf16* __restrict__ kvp,
    const __bf16* __restrict__ Vp, __bf16* __restrict__ obf)
{
  __shared__ float  S[2][16][LP_];
  __shared__ __bf16 P[2][16][LP_];
  int w = threadIdx.x >> 5, lane = threadIdx.x & 31;
  int ml = lane & 15, hi = lane >> 4;
  int gw = blockIdx.x * 2 + w;
  const int nQT = (L_ + 15) / 16;            // 13 q tiles
  int qt = gw % nQT;
  int nh = gw / nQT;
  int h = nh & (H_ - 1);
  int n = nh >> 1;
  const float scale = 0.125f;                // HD^-0.5

  float  (*Sw)[LP_] = S[w];
  __bf16 (*Pw)[LP_] = P[w];

  for (int kt = 0; kt < LP_ / 16; ++kt) {
    v8f acc = {};
    for (int kk = 0; kk < HD_; kk += 32) {
      v16bf af = {}, bf = {};
      int qrow = qt * 16 + ml;
      if (qrow < L_)
        af = load_a_frag(qp + ((size_t)(n * L_ + qrow)) * E_ + h * HD_ + kk, hi);
      int key = kt * 16 + ml;
      if (key < L_) {
        const __bf16* Kc =
            kvp + ((size_t)(n * L_ + key)) * 256 + h * HD_ + kk + hi * 16;
#pragma unroll
        for (int j = 0; j < 16; ++j) bf[j] = Kc[j];
      }
      acc = wmma_bf16(af, bf, acc);
    }
#pragma unroll
    for (int r = 0; r < 8; ++r) {
      int qrow = qt * 16 + hi * 8 + r;
      int key  = kt * 16 + ml;
      float v = acc[r] * scale;
      if (key > qrow || key >= L_) v = -1e9f;       // causal + pad mask
      Sw[hi * 8 + r][kt * 16 + ml] = v;
    }
  }
  __syncthreads();
  if (lane < 16) {
    float mx = -1e30f;
    for (int j = 0; j < LP_; ++j) mx = fmaxf(mx, Sw[lane][j]);
    float sum = 0.f;
    for (int j = 0; j < LP_; ++j) sum += __expf(Sw[lane][j] - mx);
    float inv = 1.f / sum;
    for (int j = 0; j < LP_; ++j)
      Pw[lane][j] = tobf(__expf(Sw[lane][j] - mx) * inv);
  }
  __syncthreads();
  const __bf16* Vbase = Vp + ((size_t)(n * H_ + h) * (LP_ / 32)) * 2048 + lane * 16;
  for (int ot = 0; ot < HD_ / 16; ++ot) {
    v8f acc = {};
    for (int kk = 0; kk < LP_; kk += 32) {
      v16bf af = load_a_frag(&Pw[ml][kk], hi);
      v16bf bf = load_packed_frag(Vbase + ((size_t)(kk >> 5) * 4 + ot) * 512);
      acc = wmma_bf16(af, bf, acc);
    }
#pragma unroll
    for (int r = 0; r < 8; ++r) {
      int qrow = qt * 16 + hi * 8 + r;
      if (qrow < L_) {
        size_t idx = ((size_t)(n * L_ + qrow)) * E_ + h * HD_ + ot * 16 + ml;
        obf[idx] = tobf(acc[r]);
      }
    }
  }
}

__global__ __launch_bounds__(128) void logits_kernel(
    const float* __restrict__ feats, const float* __restrict__ pos_t,
    const float* __restrict__ neg_t, float* __restrict__ out)
{
  __shared__ float red[128];
  int t = blockIdx.x, e = threadIdx.x;
  float f = feats[(size_t)t * E_ + e];
  float p  = blockReduce128(f * pos_t[(size_t)t * E_ + e], red);
  float ng = blockReduce128(f * neg_t[(size_t)t * E_ + e], red);
  if (e == 0) { out[t] = p; out[T0_ + t] = ng; }
}

__global__ void init_scalars(float* s) {
  if (threadIdx.x < 4) s[threadIdx.x] = 0.f;
}

__global__ void finalize_kernel(const float* __restrict__ s,
                                float* __restrict__ out)
{
  if (threadIdx.x == 0) {
    float nv = s[0];
    out[2 * T0_]     = s[2] / nv;   // ind_loss
    out[2 * T0_ + 1] = s[1] / nv;   // cent_loss
  }
}

// ---------------------------------------------------------------------------
static inline void launch_gemm(hipStream_t st,
    const __bf16* A, int lda, const __bf16* Bp,
    float* C, int ldc, const float* bias, const float* res, int ldres,
    __bf16* Cbf, int M, int Nn, int K, int flags)
{
  int tiles = (M / 16) * (Nn / 64);
  int blocks = (tiles + 3) / 4;
  wmma_gemm<<<blocks, 128, 0, st>>>(A, lda, Bp, C, ldc, bias, res, ldres,
                                    Cbf, M, Nn, K, flags);
}

extern "C" void kernel_launch(void* const* d_in, const int* in_sizes, int n_in,
                              void* d_out, int out_size, void* d_ws, size_t ws_size,
                              hipStream_t stream)
{
  (void)in_sizes; (void)n_in; (void)out_size; (void)ws_size;

  const float* item_emb   = (const float*)d_in[0];
  const float* pos_emb    = (const float*)d_in[1];
  const float* W_asp      = (const float*)d_in[2];
  const float* w_weight   = (const float*)d_in[3];
  const float* w_bias     = (const float*)d_in[4];
  const float* attn_ln_g  = (const float*)d_in[5];
  const float* attn_ln_b  = (const float*)d_in[6];
  const float* in_proj_w  = (const float*)d_in[7];
  const float* in_proj_b  = (const float*)d_in[8];
  const float* out_proj_w = (const float*)d_in[9];
  const float* out_proj_b = (const float*)d_in[10];
  const float* fwd_ln_g   = (const float*)d_in[11];
  const float* fwd_ln_b   = (const float*)d_in[12];
  const float* conv1_w    = (const float*)d_in[13];
  const float* conv1_b    = (const float*)d_in[14];
  const float* conv2_w    = (const float*)d_in[15];
  const float* conv2_b    = (const float*)d_in[16];
  const float* last_ln_g  = (const float*)d_in[17];
  const float* last_ln_b  = (const float*)d_in[18];
  const int*   log_seqs   = (const int*)d_in[20];
  const int*   pos_seqs   = (const int*)d_in[21];
  const int*   neg_seqs   = (const int*)d_in[22];
  float* out = (float*)d_out;

  // --- workspace carve (bump allocator, 256B aligned) ---
  char* base = (char*)d_ws;
  size_t off = 0;
  auto alloc = [&](size_t bytes) -> void* {
    void* r = base + off;
    off = (off + bytes + 255) & ~(size_t)255;
    return r;
  };
  float*  kv_save = (float*)alloc((size_t)T1_ * E_ * 4);
  __bf16* kv_bf   = (__bf16*)alloc((size_t)T1_ * E_ * 2);
  float*  validE  = (float*)alloc((size_t)T1_ * 4);
  float*  validL  = (float*)alloc((size_t)T0_ * 4);
  float*  neg_emb = (float*)alloc((size_t)T0_ * E_ * 4);
  __bf16* neg_bf  = (__bf16*)alloc((size_t)T0_ * E_ * 2);
  __bf16* Wb_asp  = (__bf16*)alloc((size_t)E_ * 2048 * 2);
  __bf16* Wmb     = (__bf16*)alloc((size_t)E_ * E_ * 2);
  float*  kv4     = (float*)alloc((size_t)T1_ * 2048 * 4);
  float*  neg4    = (float*)alloc((size_t)T0_ * 2048 * 4);
  float*  tmean   = (float*)alloc((size_t)T1_ * E_ * 4);
  float*  sbuf    = (float*)alloc((size_t)T1_ * E_ * 4);
  float*  qbuf    = (float*)alloc((size_t)T1_ * E_ * 4);
  float*  re_all  = (float*)alloc((size_t)T1_ * E_ * 4);
  float*  kvmean  = (float*)alloc((size_t)T1_ * E_ * 4);
  float*  seqs    = (float*)alloc((size_t)T0_ * E_ * 4);
  __bf16* seqs_bf = (__bf16*)alloc((size_t)T0_ * E_ * 2);
  float*  ln_buf  = (float*)alloc((size_t)T0_ * E_ * 4);
  __bf16* ln_bf   = (__bf16*)alloc((size_t)T0_ * E_ * 2);
  float*  cscr    = (float*)alloc((size_t)T0_ * 256 * 4);   // unread fp32 C scratch
  __bf16* qp_bf   = (__bf16*)alloc((size_t)T0_ * E_ * 2);
  __bf16* kvp_bf  = (__bf16*)alloc((size_t)T0_ * 256 * 2);
  __bf16* Vp      = (__bf16*)alloc((size_t)N_ * H_ * (LP_ / 32) * 2048 * 2);
  __bf16* o_bf    = (__bf16*)alloc((size_t)T0_ * E_ * 2);
  __bf16* h_bf    = (__bf16*)alloc((size_t)T0_ * E_ * 2);
  float*  pos_t   = (float*)alloc((size_t)T0_ * E_ * 4);
  float*  neg_t   = (float*)alloc((size_t)T0_ * E_ * 4);
  __bf16* Wq_p[NB_]; __bf16* Wkv_p[NB_]; __bf16* Wo_p[NB_];
  __bf16* Wc1_p[NB_]; __bf16* Wc2_p[NB_];
  for (int i = 0; i < NB_; ++i) {
    Wq_p[i]  = (__bf16*)alloc((size_t)E_ * E_ * 2);
    Wkv_p[i] = (__bf16*)alloc((size_t)E_ * 2 * E_ * 2);
    Wo_p[i]  = (__bf16*)alloc((size_t)E_ * E_ * 2);
    Wc1_p[i] = (__bf16*)alloc((size_t)E_ * E_ * 2);
    Wc2_p[i] = (__bf16*)alloc((size_t)E_ * E_ * 2);
  }
  float* scal = (float*)alloc(16 * 4);   // [0]=nvalid [1]=cent [2]=ind

  // --- pipeline ---
  init_scalars<<<1, 32, 0, stream>>>(scal);
  gather_kernel<<<(T1_ * E_ + 255) / 256, 256, 0, stream>>>(
      item_emb, log_seqs, pos_seqs, neg_seqs,
      kv_save, kv_bf, validE, validL, neg_emb, neg_bf);
  build_wasp_packed<<<(2048 * E_ + 255) / 256, 256, 0, stream>>>(W_asp, Wb_asp);
  build_wmean_packed<<<(E_ * E_ + 255) / 256, 256, 0, stream>>>(W_asp, Wmb);
  for (int i = 0; i < NB_; ++i) {
    const float* ipw = in_proj_w + (size_t)i * 3 * E_ * E_;
    pack_weight<<<(E_ * E_ + 255) / 256, 256, 0, stream>>>(ipw, Wq_p[i], E_, E_);
    pack_weight<<<(2 * E_ * E_ + 255) / 256, 256, 0, stream>>>(ipw + E_ * E_, Wkv_p[i], 2 * E_, E_);
    pack_weight<<<(E_ * E_ + 255) / 256, 256, 0, stream>>>(out_proj_w + (size_t)i * E_ * E_, Wo_p[i], E_, E_);
    pack_weight<<<(E_ * E_ + 255) / 256, 256, 0, stream>>>(conv1_w + (size_t)i * E_ * E_, Wc1_p[i], E_, E_);
    pack_weight<<<(E_ * E_ + 255) / 256, 256, 0, stream>>>(conv2_w + (size_t)i * E_ * E_, Wc2_p[i], E_, E_);
  }
  // aspect projections (fused over 16 aspects: N = 2048) + Wmean for cent
  launch_gemm(stream, kv_bf, E_, Wb_asp, kv4, 2048,
              nullptr, nullptr, 0, nullptr, T1_, 2048, E_, 0);
  launch_gemm(stream, kv_bf, E_, Wmb, tmean, E_,
              nullptr, nullptr, 0, nullptr, T1_, E_, E_, 0);
  aspect_post<<<T1_, 128, 0, stream>>>(kv_save, kv4, validE, tmean, sbuf, scal + 1, 1);
  launch_gemm(stream, neg_bf, E_, Wb_asp, neg4, 2048,
              nullptr, nullptr, 0, nullptr, T0_, 2048, E_, 0);
  aspect_post<<<T0_, 128, 0, stream>>>(neg_emb, neg4, validL, nullptr, nullptr, nullptr, 0);
  ind_kernel<<<T1_ / 8, 256, 0, stream>>>(kv4, validE, scal + 2);
  cumsum_kernel<<<N_, 128, 0, stream>>>(sbuf, validE, qbuf, scal);
  gate_kernel<<<T1_, 128, 0, stream>>>(qbuf, kv4, w_weight, w_bias, validE, re_all, kvmean, 0);
  gate_kernel<<<T0_, 128, 0, stream>>>(qbuf, neg4, w_weight, w_bias, validE, neg_t, nullptr, 1);
  build_seq<<<(T0_ * E_ + 255) / 256, 256, 0, stream>>>(
      re_all, kvmean, validE, pos_emb, seqs, seqs_bf, pos_t);

  for (int i = 0; i < NB_; ++i) {
    ln_kernel<<<T0_, 128, 0, stream>>>(seqs, attn_ln_g + i * E_, attn_ln_b + i * E_, ln_buf, ln_bf);
    launch_gemm(stream, ln_bf, E_, Wq_p[i], cscr, E_,
                in_proj_b + i * 3 * E_, nullptr, 0, qp_bf,
                T0_, E_, E_, GF_BIAS | GF_BF16);
    launch_gemm(stream, seqs_bf, E_, Wkv_p[i], cscr, 2 * E_,
                in_proj_b + i * 3 * E_ + E_, nullptr, 0, kvp_bf,
                T0_, 2 * E_, E_, GF_BIAS | GF_BF16);
    pack_v_kernel<<<(N_ * H_ * (LP_ / 32) * 2048 + 255) / 256, 256, 0, stream>>>(kvp_bf, Vp);
    attn_kernel<<<(N_ * H_ * 13) / 2, 64, 0, stream>>>(qp_bf, kvp_bf, Vp, o_bf);
    launch_gemm(stream, o_bf, E_, Wo_p[i], seqs, E_,
                out_proj_b + i * E_, ln_buf, E_, nullptr,
                T0_, E_, E_, GF_BIAS | GF_RES);
    ln_kernel<<<T0_, 128, 0, stream>>>(seqs, fwd_ln_g + i * E_, fwd_ln_b + i * E_, ln_buf, ln_bf);
    launch_gemm(stream, ln_bf, E_, Wc1_p[i], cscr, E_,
                conv1_b + i * E_, nullptr, 0, h_bf,
                T0_, E_, E_, GF_BIAS | GF_RELU | GF_BF16);
    launch_gemm(stream, h_bf, E_, Wc2_p[i], seqs, E_,
                conv2_b + i * E_, ln_buf, E_, nullptr,
                T0_, E_, E_, GF_BIAS | GF_RES);
    mask_bf_kernel<<<(T0_ * E_ + 255) / 256, 256, 0, stream>>>(seqs, seqs_bf, validE);
  }

  ln_kernel<<<T0_, 128, 0, stream>>>(seqs, last_ln_g, last_ln_b, ln_buf, nullptr);
  logits_kernel<<<T0_, 128, 0, stream>>>(ln_buf, pos_t, neg_t, out);
  finalize_kernel<<<1, 32, 0, stream>>>(scal, out);
}
